// EMLayer_16080357556717
// MI455X (gfx1250) — matrix operand order repaired and checked by reading, not compiled, not run on hardware
//
#include <hip/hip_runtime.h>

// ---------------------------------------------------------------------------
// Problem constants (from reference setup_inputs)
// ---------------------------------------------------------------------------
#define NPTS   524288      // N points
#define MVOX   131072      // M voxels
#define BATCH  2
#define GDIM   64
#define NGRID  (BATCH * GDIM * GDIM * GDIM)   // 524288 grid sites
#define EPSBN  1e-5f

typedef _Float16 f16;
typedef __attribute__((ext_vector_type(16))) _Float16 v16h;
typedef __attribute__((ext_vector_type(8)))  _Float16 v8h;
typedef __attribute__((ext_vector_type(8)))  float    v8f;
typedef __attribute__((ext_vector_type(4)))  int      v4i;

// ---------------------------------------------------------------------------
// CDNA5 async global->LDS copy (ASYNCcnt-tracked), with portable fallback.
// Builtin signature (from compiler diagnostic): (v4i AS1*, v4i AS3*, Ii, Ii)
// ---------------------------------------------------------------------------
#if defined(__has_builtin)
#  if __has_builtin(__builtin_amdgcn_global_load_async_to_lds_b128)
#    define HAVE_ASYNC_LDS 1
#  endif
#endif
#ifndef HAVE_ASYNC_LDS
#  define HAVE_ASYNC_LDS 0
#endif

#if HAVE_ASYNC_LDS
typedef __attribute__((address_space(1))) v4i gbl_v4i;
typedef __attribute__((address_space(3))) v4i lds_v4i;
__device__ __forceinline__ void async_copy16(void* lds_dst, const void* gsrc) {
  // emits global_load_async_to_lds_b128 (tracked by ASYNCcnt)
  __builtin_amdgcn_global_load_async_to_lds_b128((gbl_v4i*)gsrc,
                                                 (lds_v4i*)lds_dst, 0, 0);
}
__device__ __forceinline__ void wait_async0() {
#  if __has_builtin(__builtin_amdgcn_s_wait_asynccnt)
  __builtin_amdgcn_s_wait_asynccnt(0);
#  else
  asm volatile("s_wait_asynccnt 0x0" ::: "memory");
#  endif
}
#else
__device__ __forceinline__ void async_copy16(void* lds_dst, const void* gsrc) {
  *(uint4*)lds_dst = *(const uint4*)gsrc;
}
__device__ __forceinline__ void wait_async0() {}
#endif

// ---------------------------------------------------------------------------
// Workspace layout (bytes). [grid | pooled | cnt | stats] form the zero span.
// ---------------------------------------------------------------------------
#define OFF_GRID    ((size_t)0)                         // NGRID*32 f16 = 33,554,432
#define OFF_POOLED  ((size_t)33554432)                  // M*32 f32    = 16,777,216
#define OFF_CNT     ((size_t)50331648)                  // M f32       =    524,288
#define OFF_STATS   ((size_t)50855936)                  // 6*64 f32    =      1,536
#define ZERO_BYTES  ((size_t)50857472)                  // zero-fill span
#define ZERO_U4     (ZERO_BYTES / 16)                   // 3,178,592 uint4
#define OFF_SS      ((size_t)50857472)                  // 6*64 f32 scale/shift
#define OFF_LIN     ((size_t)50859008)                  // M i32
#define OFF_WPC     ((size_t)51383296)                  // 27*4*512 f16 = 110,592
#define OFF_W1P     ((size_t)51493888)                  // 1*4*512 f16
#define OFF_W2P     ((size_t)51497984)                  // 2*4*512 f16
#define OFF_WLP     ((size_t)51506176)                  // 3*4*512 f16
#define OFF_DENSE   ((size_t)51518464)                  // NGRID*64 f16 = 67,108,864
#define OFF_T1      ((size_t)118627328)                 // N*64 f16
#define OFF_T2      ((size_t)185736192)                 // N*64 f16

#define CONV_W_LDS  110592                              // packed conv weights in LDS
#define CONV_A_LDS  (9 * 66 * 32 * 2)                   // 38,016
#define CONV_SMEM   (CONV_W_LDS + CONV_A_LDS)           // 148,608 (2 blocks / WGP)

// ---------------------------------------------------------------------------
// WMMA helpers (CDNA5 16x16x32 f16 -> f32)
// ---------------------------------------------------------------------------
__device__ __forceinline__ v8f wmma_f16(v16h a, v16h b, v8f c) {
  return __builtin_amdgcn_wmma_f32_16x16x32_f16(false, a, false, b,
                                                (short)0, c, false, false);
}
__device__ __forceinline__ v16h make_a(v8h lo, v8h hi) {
  return __builtin_shufflevector(lo, hi, 0, 1, 2, 3, 4, 5, 6, 7,
                                         8, 9, 10, 11, 12, 13, 14, 15);
}
__device__ __forceinline__ v8h cvt8f(const float* p) {
  v8h r;
#pragma unroll
  for (int i = 0; i < 8; ++i) r[i] = (_Float16)p[i];
  return r;
}
__device__ __forceinline__ v8h bn_relu8(const f16* p, const float* sc, const float* sh) {
  v8h r;
#pragma unroll
  for (int i = 0; i < 8; ++i) {
    float v = (float)p[i] * sc[i] + sh[i];
    r[i] = (_Float16)fmaxf(v, 0.0f);
  }
  return r;
}

// ---------------------------------------------------------------------------
// 0) zero-fill workspace span
// ---------------------------------------------------------------------------
__global__ void zero_ws_kernel(uint4* __restrict__ p, unsigned n16) {
  unsigned i = blockIdx.x * 256u + threadIdx.x;
  if (i < n16) p[i] = make_uint4(0u, 0u, 0u, 0u);
}

// ---------------------------------------------------------------------------
// 1) pack all weight matrices into WMMA B-fragment layout (f32 -> f16)
//    B frag: lane l, elem e -> k = 32*j + 8*(l>>4) + (e&7) + 16*(e>>3), n = l&15
// ---------------------------------------------------------------------------
__global__ void pack_weights_kernel(const float* __restrict__ Wc,
                                    const float* __restrict__ W1,
                                    const float* __restrict__ W2,
                                    const float* __restrict__ Wl,
                                    f16* __restrict__ wpc, f16* __restrict__ w1p,
                                    f16* __restrict__ w2p, f16* __restrict__ wlp) {
  int idx = blockIdx.x * 256 + threadIdx.x;
  if (idx < 55296) {                      // conv: 27 taps x 4 ct x 512
    int tap = idx / 2048, rem = idx % 2048;
    int ct = rem / 512, q = rem % 512;
    int l = q / 16, e = q % 16;
    int k = 8 * (l >> 4) + (e & 7) + 16 * (e >> 3);  // ci
    int co = ct * 16 + (l & 15);
    wpc[idx] = (f16)Wc[(co * 32 + k) * 27 + tap];
  } else if (idx < 55296 + 2048) {        // W1: K=32 (j=0)
    int q2 = idx - 55296;
    int ct = q2 / 512, q = q2 % 512, l = q / 16, e = q % 16;
    int k = 8 * (l >> 4) + (e & 7) + 16 * (e >> 3);
    int co = ct * 16 + (l & 15);
    w1p[q2] = (f16)W1[k * 64 + co];
  } else if (idx < 55296 + 2048 + 4096) { // W2: K=64 (j=0..1)
    int q2 = idx - 57344;
    int j = q2 / 2048, rem = q2 % 2048;
    int ct = rem / 512, q = rem % 512, l = q / 16, e = q % 16;
    int k = 32 * j + 8 * (l >> 4) + (e & 7) + 16 * (e >> 3);
    int co = ct * 16 + (l & 15);
    w2p[q2] = (f16)W2[k * 64 + co];
  } else if (idx < 67584) {               // W_lin: K=96 (j=0..2)
    int q2 = idx - 61440;
    int j = q2 / 2048, rem = q2 % 2048;
    int ct = rem / 512, q = rem % 512, l = q / 16, e = q % 16;
    int k = 32 * j + 8 * (l >> 4) + (e & 7) + 16 * (e >> 3);
    int co = ct * 16 + (l & 15);
    wlp[q2] = (f16)Wl[k * 64 + co];
  }
}

// ---------------------------------------------------------------------------
// 2) scatter-add points into voxel sums + counts
// ---------------------------------------------------------------------------
__global__ void scatter_add_kernel(const float* __restrict__ x,
                                   const int* __restrict__ cluster,
                                   float* __restrict__ pooled,
                                   float* __restrict__ cnt) {
  unsigned tid = blockIdx.x * 256u + threadIdx.x;   // N*32 threads exactly
  int i = tid >> 5, c = tid & 31;
  int cl = cluster[i];
  atomicAdd(&pooled[cl * 32 + c], x[tid]);
  if (c == 0) atomicAdd(&cnt[cl], 1.0f);
}

// ---------------------------------------------------------------------------
// 3) pooled/cnt -> f16 grid at lin; record lin per voxel
// ---------------------------------------------------------------------------
__global__ void voxelize_kernel(const float* __restrict__ pooled,
                                const float* __restrict__ cnt,
                                const int* __restrict__ batch,
                                const int* __restrict__ upi,
                                const int* __restrict__ coords,
                                f16* __restrict__ grid, int* __restrict__ linbuf) {
  unsigned tid = blockIdx.x * 256u + threadIdx.x;   // M*32 threads exactly
  int i = tid >> 5, c = tid & 31;
  int vb = batch[upi[i]];
  int c0 = coords[i * 3 + 0], c1 = coords[i * 3 + 1], c2 = coords[i * 3 + 2];
  int lin = ((vb * GDIM + c0) * GDIM + c1) * GDIM + c2;
  float v = pooled[tid] / fmaxf(cnt[i], 1.0f);
  grid[lin * 32 + c] = (f16)v;
  if (c == 0) linbuf[i] = lin;
}

// ---------------------------------------------------------------------------
// 4) dense 3x3x3 conv as implicit WMMA GEMM.
//    Block = 128 thr = 4 waves; one block per (b,z,y) row; wave w -> x-tile w.
//    Dynamic LDS: [packed weights 110,592 B][activation window 38,016 B].
//    Staging uses global_load_async_to_lds_b128 when available.
// ---------------------------------------------------------------------------
__global__ void __launch_bounds__(128)
conv3d_wmma_kernel(const f16* __restrict__ grid, const f16* __restrict__ wpack,
                   f16* __restrict__ dense) {
  extern __shared__ __align__(16) char smem[];
  f16* wlds = (f16*)smem;                                       // [27*4*512]
  _Float16 (*alds)[66][32] = (_Float16 (*)[66][32])(smem + CONV_W_LDS);

  int row = blockIdx.x;                 // 0..8191 : b*4096 + z*64 + y
  int b = row >> 12;
  int z = (row >> 6) & 63;
  int y = row & 63;
  int t = threadIdx.x;

  // ---- stage packed conv weights: 6912 x 16B chunks ----
  for (int idx = t; idx < 6912; idx += 128)
    async_copy16(wlds + idx * 8, wpack + idx * 8);

  // ---- stage activation window: 9 segs x 66 rows x 4 chunks of 8 f16 ----
  for (int idx = t; idx < 9 * 66 * 4; idx += 128) {
    int seg = idx / 264;                // 66*4
    int r = (idx >> 2) % 66;
    int c4 = idx & 3;
    int zz = z + seg / 3 - 1;
    int yy = y + seg % 3 - 1;
    int xx = r - 1;
    void* dst = &alds[seg][r][c4 * 8];
    if ((unsigned)zz < 64u && (unsigned)yy < 64u && (unsigned)xx < 64u) {
      async_copy16(dst, grid + ((((b * 64 + zz) * 64 + yy) * 64 + xx) * 32 + c4 * 8));
    } else {
      *(uint4*)dst = make_uint4(0u, 0u, 0u, 0u);   // disjoint from async targets
    }
  }
  wait_async0();
  __syncthreads();

  int wave = t >> 5, lane = t & 31, m = lane & 15, half = lane >> 4;
  v8f acc[4] = {};
  for (int tap = 0; tap < 27; ++tap) {
    int seg = tap / 3;                  // kd*3+kh
    int dx = tap % 3 - 1;               // kw-1
    const _Float16* rp = &alds[seg][wave * 16 + m + 1 + dx][half * 8];
    v16h a = make_a(*(const v8h*)rp, *(const v8h*)(rp + 16));
#pragma unroll
    for (int ct = 0; ct < 4; ++ct) {
      v16h bf = *(const v16h*)(wlds + (tap * 4 + ct) * 512 + lane * 16);
      acc[ct] = wmma_f16(a, bf, acc[ct]);
    }
  }
  int sbase = row * 64 + wave * 16;     // site index base
#pragma unroll
  for (int ct = 0; ct < 4; ++ct)
#pragma unroll
    for (int r = 0; r < 8; ++r)
      dense[(sbase + r + 8 * half) * 64 + ct * 16 + m] = (_Float16)acc[ct][r];
}

// ---------------------------------------------------------------------------
// 5) GEMM1: t1 = f16(x)[Nx32] @ W1 + b1 ; fused column sum/sumsq
//    Block = 256 thr = 8 waves; wave -> 16-row tile. Weights staged to LDS.
// ---------------------------------------------------------------------------
__global__ void __launch_bounds__(256)
gemm1_kernel(const float* __restrict__ x, const f16* __restrict__ w1p,
             const float* __restrict__ b1, f16* __restrict__ t1,
             float* __restrict__ sum, float* __restrict__ sumsq) {
  __shared__ float ssum[64], ssq[64];
  __shared__ __align__(16) f16 wsh[2048];
  int t = threadIdx.x;
  if (t < 64) { ssum[t] = 0.f; ssq[t] = 0.f; }
  if (t < 128) async_copy16(wsh + t * 8, w1p + t * 8);   // 128 x 16B
  wait_async0();
  __syncthreads();
  int wave = t >> 5, lane = t & 31, m = lane & 15, half = lane >> 4;
  int tile = blockIdx.x * 8 + wave;
  const float* xr = x + (tile * 16 + m) * 32 + half * 8;
  v16h a = make_a(cvt8f(xr), cvt8f(xr + 16));
  v8f acc[4] = {};
#pragma unroll
  for (int ct = 0; ct < 4; ++ct) {
    v16h bf = *(const v16h*)(wsh + ct * 512 + lane * 16);
    acc[ct] = wmma_f16(a, bf, acc[ct]);
  }
#pragma unroll
  for (int ct = 0; ct < 4; ++ct) {
    int col = ct * 16 + m;
    float bias = b1[col];
    float ps = 0.f, pq = 0.f;
#pragma unroll
    for (int r = 0; r < 8; ++r) {
      float v = acc[ct][r] + bias;
      int rr = tile * 16 + r + 8 * half;
      t1[rr * 64 + col] = (f16)v;
      ps += v; pq += v * v;
    }
    atomicAdd(&ssum[col], ps);
    atomicAdd(&ssq[col], pq);
  }
  __syncthreads();
  if (t < 64) { atomicAdd(&sum[t], ssum[t]); atomicAdd(&sumsq[t], ssq[t]); }
}

// ---------------------------------------------------------------------------
// 6) per-channel BN coefficients from accumulated stats
// ---------------------------------------------------------------------------
__global__ void finalize_stats_kernel(const float* __restrict__ sum,
                                      const float* __restrict__ sumsq,
                                      const float* __restrict__ g,
                                      const float* __restrict__ beta,
                                      float* __restrict__ scale,
                                      float* __restrict__ shift, float invN) {
  int c = threadIdx.x;
  float mean = sum[c] * invN;
  float var = sumsq[c] * invN - mean * mean;
  float s = g[c] * rsqrtf(var + EPSBN);
  scale[c] = s;
  shift[c] = beta[c] - mean * s;
}

// ---------------------------------------------------------------------------
// 7) GEMM2: t2 = relu(bn1(t1)) @ W2 + b2 ; fused stats (K=64)
// ---------------------------------------------------------------------------
__global__ void __launch_bounds__(256)
gemm2_kernel(const f16* __restrict__ t1, const float* __restrict__ sc1,
             const float* __restrict__ sh1, const f16* __restrict__ w2p,
             const float* __restrict__ b2, f16* __restrict__ t2,
             float* __restrict__ sum, float* __restrict__ sumsq) {
  __shared__ float ssum[64], ssq[64];
  __shared__ __align__(16) f16 wsh[4096];
  int t = threadIdx.x;
  if (t < 64) { ssum[t] = 0.f; ssq[t] = 0.f; }
  async_copy16(wsh + t * 8, w2p + t * 8);                // 256 x 16B
  wait_async0();
  __syncthreads();
  int wave = t >> 5, lane = t & 31, m = lane & 15, half = lane >> 4;
  int tile = blockIdx.x * 8 + wave;
  const f16* arow = t1 + (tile * 16 + m) * 64;
  v8f acc[4] = {};
#pragma unroll
  for (int j = 0; j < 2; ++j) {
    int k0 = 32 * j + 8 * half;
    v16h a = make_a(bn_relu8(arow + k0, sc1 + k0, sh1 + k0),
                    bn_relu8(arow + k0 + 16, sc1 + k0 + 16, sh1 + k0 + 16));
#pragma unroll
    for (int ct = 0; ct < 4; ++ct) {
      v16h bf = *(const v16h*)(wsh + (j * 4 + ct) * 512 + lane * 16);
      acc[ct] = wmma_f16(a, bf, acc[ct]);
    }
  }
#pragma unroll
  for (int ct = 0; ct < 4; ++ct) {
    int col = ct * 16 + m;
    float bias = b2[col];
    float ps = 0.f, pq = 0.f;
#pragma unroll
    for (int r = 0; r < 8; ++r) {
      float v = acc[ct][r] + bias;
      int rr = tile * 16 + r + 8 * half;
      t2[rr * 64 + col] = (f16)v;
      ps += v; pq += v * v;
    }
    atomicAdd(&ssum[col], ps);
    atomicAdd(&ssq[col], pq);
  }
  __syncthreads();
  if (t < 64) { atomicAdd(&sum[t], ssum[t]); atomicAdd(&sumsq[t], ssq[t]); }
}

// ---------------------------------------------------------------------------
// 8) GEMM3: s = [x | convF[cluster]] @ W_lin + b_lin + relu(bn2(t2)) ; stats
//    K = 96 = 32 (x) + 64 (gathered conv features).
// ---------------------------------------------------------------------------
__global__ void __launch_bounds__(256)
gemm3_kernel(const float* __restrict__ x, const f16* __restrict__ dense,
             const int* __restrict__ cluster, const int* __restrict__ linbuf,
             const f16* __restrict__ wlp, const float* __restrict__ b_lin,
             const f16* __restrict__ t2, const float* __restrict__ sc2,
             const float* __restrict__ sh2, float* __restrict__ sout,
             float* __restrict__ sum, float* __restrict__ sumsq) {
  __shared__ float ssum[64], ssq[64];
  __shared__ __align__(16) f16 wsh[6144];
  int t = threadIdx.x;
  if (t < 64) { ssum[t] = 0.f; ssq[t] = 0.f; }
  for (int idx = t; idx < 384; idx += 256)               // 384 x 16B
    async_copy16(wsh + idx * 8, wlp + idx * 8);
  wait_async0();
  __syncthreads();
  int wave = t >> 5, lane = t & 31, m = lane & 15, half = lane >> 4;
  int tile = blockIdx.x * 8 + wave;
  int row = tile * 16 + m;
  const f16* fr = dense + linbuf[cluster[row]] * 64;
  v8f acc[4] = {};
  {  // j=0 : point features x
    const float* xr = x + row * 32 + half * 8;
    v16h a = make_a(cvt8f(xr), cvt8f(xr + 16));
#pragma unroll
    for (int ct = 0; ct < 4; ++ct) {
      v16h bf = *(const v16h*)(wsh + ct * 512 + lane * 16);
      acc[ct] = wmma_f16(a, bf, acc[ct]);
    }
  }
#pragma unroll
  for (int j = 1; j < 3; ++j) {  // conv-feature channels [0,32) and [32,64)
    const f16* p = fr + 32 * (j - 1) + 8 * half;
    v16h a = make_a(*(const v8h*)p, *(const v8h*)(p + 16));
#pragma unroll
    for (int ct = 0; ct < 4; ++ct) {
      v16h bf = *(const v16h*)(wsh + (j * 4 + ct) * 512 + lane * 16);
      acc[ct] = wmma_f16(a, bf, acc[ct]);
    }
  }
#pragma unroll
  for (int ct = 0; ct < 4; ++ct) {
    int col = ct * 16 + m;
    float bias = b_lin[col];
    float sc = sc2[col], sh = sh2[col];
    float ps = 0.f, pq = 0.f;
#pragma unroll
    for (int r = 0; r < 8; ++r) {
      int rr = tile * 16 + r + 8 * half;
      float h2 = fmaxf((float)t2[rr * 64 + col] * sc + sh, 0.0f);
      float v = acc[ct][r] + bias + h2;
      sout[rr * 64 + col] = v;
      ps += v; pq += v * v;
    }
    atomicAdd(&ssum[col], ps);
    atomicAdd(&ssq[col], pq);
  }
  __syncthreads();
  if (t < 64) { atomicAdd(&sum[t], ssum[t]); atomicAdd(&sumsq[t], ssq[t]); }
}

// ---------------------------------------------------------------------------
// 9) final: out = relu(bn3(s)) in place on d_out
// ---------------------------------------------------------------------------
__global__ void final_apply_kernel(float* __restrict__ out,
                                   const float* __restrict__ scale,
                                   const float* __restrict__ shift) {
  unsigned i = blockIdx.x * 256u + threadIdx.x;   // N*64 threads exactly
  int c = i & 63;
  out[i] = fmaxf(out[i] * scale[c] + shift[c], 0.0f);
}

// ---------------------------------------------------------------------------
extern "C" void kernel_launch(void* const* d_in, const int* in_sizes, int n_in,
                              void* d_out, int out_size, void* d_ws, size_t ws_size,
                              hipStream_t stream) {
  const float* x      = (const float*)d_in[0];
  const float* W1     = (const float*)d_in[1];
  const float* b1     = (const float*)d_in[2];
  const float* g1     = (const float*)d_in[3];
  const float* beta1  = (const float*)d_in[4];
  const float* W2     = (const float*)d_in[5];
  const float* b2     = (const float*)d_in[6];
  const float* g2     = (const float*)d_in[7];
  const float* beta2  = (const float*)d_in[8];
  const float* W_conv = (const float*)d_in[9];
  const float* W_lin  = (const float*)d_in[10];
  const float* b_lin  = (const float*)d_in[11];
  const float* gamma  = (const float*)d_in[12];
  const float* betaF  = (const float*)d_in[13];
  const int* cluster  = (const int*)d_in[14];
  const int* batch    = (const int*)d_in[15];
  const int* upi      = (const int*)d_in[16];
  const int* coords   = (const int*)d_in[17];

  char* ws = (char*)d_ws;
  f16*   grid   = (f16*)  (ws + OFF_GRID);
  float* pooled = (float*)(ws + OFF_POOLED);
  float* cnt    = (float*)(ws + OFF_CNT);
  float* stats  = (float*)(ws + OFF_STATS);   // 6 x 64
  float* ss     = (float*)(ws + OFF_SS);      // 6 x 64
  int*   linbuf = (int*)  (ws + OFF_LIN);
  f16*   wpc    = (f16*)  (ws + OFF_WPC);
  f16*   w1p    = (f16*)  (ws + OFF_W1P);
  f16*   w2p    = (f16*)  (ws + OFF_W2P);
  f16*   wlp    = (f16*)  (ws + OFF_WLP);
  f16*   dense  = (f16*)  (ws + OFF_DENSE);
  f16*   t1     = (f16*)  (ws + OFF_T1);
  f16*   t2     = (f16*)  (ws + OFF_T2);
  float* out    = (float*)d_out;
  const float invN = 1.0f / (float)NPTS;

  zero_ws_kernel<<<(unsigned)((ZERO_U4 + 255) / 256), 256, 0, stream>>>(
      (uint4*)ws, (unsigned)ZERO_U4);
  pack_weights_kernel<<<(67584 + 255) / 256, 256, 0, stream>>>(
      W_conv, W1, W2, W_lin, wpc, w1p, w2p, wlp);
  scatter_add_kernel<<<NPTS * 32 / 256, 256, 0, stream>>>(x, cluster, pooled, cnt);
  voxelize_kernel<<<MVOX * 32 / 256, 256, 0, stream>>>(
      pooled, cnt, batch, upi, coords, grid, linbuf);
  conv3d_wmma_kernel<<<BATCH * GDIM * GDIM, 128, CONV_SMEM, stream>>>(
      grid, wpc, dense);

  gemm1_kernel<<<NPTS / 128, 256, 0, stream>>>(x, w1p, b1, t1, stats, stats + 64);
  finalize_stats_kernel<<<1, 64, 0, stream>>>(stats, stats + 64, g1, beta1,
                                              ss, ss + 64, invN);
  gemm2_kernel<<<NPTS / 128, 256, 0, stream>>>(t1, ss, ss + 64, w2p, b2, t2,
                                               stats + 128, stats + 192);
  finalize_stats_kernel<<<1, 64, 0, stream>>>(stats + 128, stats + 192, g2, beta2,
                                              ss + 128, ss + 192, invN);
  gemm3_kernel<<<NPTS / 128, 256, 0, stream>>>(x, dense, cluster, linbuf, wlp, b_lin,
                                               t2, ss + 128, ss + 192, out,
                                               stats + 256, stats + 320);
  finalize_stats_kernel<<<1, 64, 0, stream>>>(stats + 256, stats + 320, gamma, betaF,
                                              ss + 256, ss + 320, invN);
  final_apply_kernel<<<NPTS * 64 / 256, 256, 0, stream>>>(out, ss + 256, ss + 320);
}
